// Attention_52261162058028
// MI455X (gfx1250) — matrix operand order repaired and checked by reading, not compiled, not run on hardware
//
#include <hip/hip_runtime.h>
#include <hip/hip_bf16.h>

// ---------------- problem constants ----------------
#define BB   2
#define SS   2048
#define HH   2048
#define NHH  16
#define HDD  128          // head dim
#define H3   (3*HH)       // 6144
#define SCALE 0.08838834764831845f   // 1/sqrt(128)

typedef __bf16 bf16;
typedef bf16  v16bf __attribute__((ext_vector_type(16)));
typedef float v8f   __attribute__((ext_vector_type(8)));
typedef unsigned int v4u __attribute__((ext_vector_type(4)));

// ---------------- helpers ----------------
__device__ __forceinline__ bf16 f2b(float f) {
  union { float f; unsigned u; } v; v.f = f;
  unsigned r = (v.u + 0x7FFFu + ((v.u >> 16) & 1u)) >> 16;   // RNE
  union { unsigned short s; bf16 b; } o; o.s = (unsigned short)r;
  return o.b;
}

// A-matrix fragment (16x32 bf16, MxK).  ISA layout:
// lane: m = lane&15, koff = (lane>=16)?8:0
// elems 0..7  = K  koff..koff+7      (VGPR0..3)
// elems 8..15 = K  16+koff..16+koff+7 (VGPR4..7)
__device__ __forceinline__ v16bf load_frag_a(const bf16* __restrict__ rows,
                                             int ld, int kbase, int lane) {
  int m    = lane & 15;
  int koff = (lane >> 4) << 3;          // 0 or 8
  const bf16* p = rows + (size_t)m * ld + kbase + koff;
  v16bf r;
  *(v4u*)&r       = *(const v4u*)p;          // 8 bf16
  *((v4u*)&r + 1) = *(const v4u*)(p + 16);   // 8 bf16
  return r;
}

// B-matrix fragment (32x16 bf16, KxN) where memory holds rowsN[n][k]
// (K-contiguous per column n).  ISA layout:
// lane: n = lane&15, kstart = (lane>=16)?8:0, elems 0..7 = K kstart..kstart+7, elems 8..15 = K 16+kstart..16+kstart+7
__device__ __forceinline__ v16bf load_frag_b(const bf16* __restrict__ rowsN,
                                             int ld, int kbase, int lane) {
  int n      = lane & 15;
  int kstart = (lane >> 4) << 3;        // 0 or 8
  const bf16* p = rowsN + (size_t)n * ld + kbase + kstart;
  v16bf r;
  *(v4u*)&r       = *(const v4u*)p;
  *((v4u*)&r + 1) = *(const v4u*)(p + 16);
  return r;
}

__device__ __forceinline__ v8f wmma_bf16(v16bf a, v16bf b, v8f c) {
  return __builtin_amdgcn_wmma_f32_16x16x32_bf16(false, a, false, b,
                                                 (short)0, c, false, false);
}

// ---------------- kernel 1: f32 -> bf16 convert (x4 vectorized) ----------------
__global__ void k_convert4(const float* __restrict__ src, bf16* __restrict__ dst,
                           size_t n4) {
  size_t i = (size_t)blockIdx.x * blockDim.x + threadIdx.x;
  if (i >= n4) return;
  float4 f = ((const float4*)src)[i];
  union { bf16 b[4]; uint2 u; } o;
  o.b[0] = f2b(f.x); o.b[1] = f2b(f.y); o.b[2] = f2b(f.z); o.b[3] = f2b(f.w);
  ((uint2*)dst)[i] = o.u;
}

// ---------------- kernel 2/5: bf16 GEMM, C = A * W^T ----------------
// A: [M][K] row-major bf16, W: [N][K] row-major bf16 (weights, out x in)
// block: 256 thr (8 waves). wave tile: 32(M) x 64(N). block tile: 128 x 128.
template <typename OutT>
__global__ __launch_bounds__(256)
void k_gemm(const bf16* __restrict__ A, const bf16* __restrict__ W,
            OutT* __restrict__ C, int M, int N, int K) {
  const int lane  = threadIdx.x & 31;
  const int wave  = threadIdx.x >> 5;
  const int mbase = blockIdx.y * 128 + (wave & 3) * 32;
  const int nbase = blockIdx.x * 128 + (wave >> 2) * 64;

  const bf16* Arows0 = A + (size_t)mbase * K;
  const bf16* Arows1 = A + (size_t)(mbase + 16) * K;

  v8f acc[2][4];
#pragma unroll
  for (int r = 0; r < 2; ++r)
#pragma unroll
    for (int t = 0; t < 4; ++t) acc[r][t] = v8f{0};

  for (int k = 0; k < K; k += 32) {
    v16bf a0 = load_frag_a(Arows0, K, k, lane);
    v16bf a1 = load_frag_a(Arows1, K, k, lane);
    if (k + 64 < K) {   // emits global_prefetch_b8 for next iterations
      __builtin_prefetch(Arows0 + (size_t)(lane & 15) * K + k + 64, 0, 0);
      __builtin_prefetch(W + (size_t)(nbase + (lane & 31)) * K + k + 64, 0, 0);
      __builtin_prefetch(W + (size_t)(nbase + 32 + (lane & 31)) * K + k + 64, 0, 0);
    }
#pragma unroll
    for (int t = 0; t < 4; ++t) {
      v16bf b = load_frag_b(W + (size_t)(nbase + 16 * t) * K, K, k, lane);
      acc[0][t] = wmma_bf16(a0, b, acc[0][t]);
      acc[1][t] = wmma_bf16(a1, b, acc[1][t]);
    }
  }

  const int n  = lane & 15;
  const int hi = (lane >> 4) << 3;      // row offset 0 or 8
#pragma unroll
  for (int r = 0; r < 2; ++r)
#pragma unroll
    for (int t = 0; t < 4; ++t)
#pragma unroll
      for (int j = 0; j < 8; ++j) {
        size_t idx = (size_t)(mbase + 16 * r + hi + j) * N + nbase + 16 * t + n;
        float v = acc[r][t][j];
        if constexpr (__is_same(OutT, bf16)) C[idx] = f2b(v);
        else                                 C[idx] = v;
      }
}

// ---------------- kernel 3: V transpose per head (LDS-tiled, coalesced) ------
// vT[b][h][d][s] = qkv[b*S+s][2H + h*HD + d]
// grid: (S/32, HD/32, B*NH), block 256 (32x8)
__global__ __launch_bounds__(256)
void k_transpose_v(const bf16* __restrict__ qkv, bf16* __restrict__ vT) {
  __shared__ bf16 tile[32][33];
  const int bh = blockIdx.z;
  const int b = bh / NHH, h = bh % NHH;
  const int s0 = blockIdx.x * 32;
  const int d0 = blockIdx.y * 32;
  const int tx = threadIdx.x & 31;
  const int ty = threadIdx.x >> 5;      // 0..7

#pragma unroll
  for (int r = ty; r < 32; r += 8)      // read: d-contiguous (coalesced)
    tile[r][tx] = qkv[(size_t)(b * SS + s0 + r) * H3 + 2 * HH +
                      (size_t)h * HDD + d0 + tx];
  __syncthreads();
#pragma unroll
  for (int r = ty; r < 32; r += 8)      // write: s-contiguous (coalesced)
    vT[((size_t)bh * HDD + d0 + r) * SS + s0 + tx] = tile[tx][r];
}

// ---------------- kernel 4: causal flash attention ----------------
// grid: (B*NH, S/128), block 256 (8 waves, each wave owns 16 query rows)
__global__ __launch_bounds__(256)
void k_flash_attn(const bf16* __restrict__ qkv, const bf16* __restrict__ vT,
                  bf16* __restrict__ y) {
  const int lane = threadIdx.x & 31;
  const int wave = threadIdx.x >> 5;
  const int bh = blockIdx.x;
  const int b = bh / NHH, h = bh % NHH;
  const int q0 = blockIdx.y * 128 + wave * 16;

  __shared__ __align__(16) bf16 pbuf[8][16 * 32];   // per-wave P tile (16x32)
  bf16* myp = &pbuf[wave][0];

  const int hi   = lane >> 4;      // half-wave id
  const int ncol = lane & 15;

  // Q fragments (entire 16x128 row block), ld = 3H
  const bf16* qrows = qkv + (size_t)(b * SS + q0) * H3 + (size_t)h * HDD;
  v16bf qf[4];
#pragma unroll
  for (int kk = 0; kk < 4; ++kk) qf[kk] = load_frag_a(qrows, H3, kk * 32, lane);

  const bf16* krows = qkv + (size_t)b * SS * H3 + HH + (size_t)h * HDD;
  const bf16* vtb   = vT + (size_t)bh * HDD * SS;

  v8f o[8];
#pragma unroll
  for (int t = 0; t < 8; ++t) o[t] = v8f{0};
  float mrun[8], lrun[8];
#pragma unroll
  for (int j = 0; j < 8; ++j) { mrun[j] = -1e30f; lrun[j] = 0.0f; }

  const int nkb = (q0 + 15) / 32 + 1;       // causal: key blocks of 32
  for (int kb = 0; kb < nkb; ++kb) {
    const int key0 = kb * 32;

    // ---- S = Q*K^T (two 16x16 tiles over 32 keys) ----
    v8f s0 = v8f{0}, s1 = v8f{0};
#pragma unroll
    for (int kk = 0; kk < 4; ++kk) {
      v16bf b0 = load_frag_b(krows + (size_t)key0 * H3, H3, kk * 32, lane);
      v16bf b1 = load_frag_b(krows + (size_t)(key0 + 16) * H3, H3, kk * 32, lane);
      s0 = wmma_bf16(qf[kk], b0, s0);
      s1 = wmma_bf16(qf[kk], b1, s1);
    }

    // ---- scale, causal mask, online softmax ----
#pragma unroll
    for (int j = 0; j < 8; ++j) {
      const int qrow = q0 + hi * 8 + j;
      float a0 = s0[j] * SCALE;
      float a1 = s1[j] * SCALE;
      if (key0 + ncol > qrow)      a0 = -1e30f;
      if (key0 + 16 + ncol > qrow) a1 = -1e30f;

      float mx = fmaxf(a0, a1);
#pragma unroll
      for (int d = 1; d < 16; d <<= 1) mx = fmaxf(mx, __shfl_xor(mx, d, 32));
      const float mnew = fmaxf(mrun[j], mx);
      const float corr = __expf(mrun[j] - mnew);
      const float p0 = __expf(a0 - mnew);
      const float p1 = __expf(a1 - mnew);
      float rs = p0 + p1;
#pragma unroll
      for (int d = 1; d < 16; d <<= 1) rs += __shfl_xor(rs, d, 32);
      lrun[j] = lrun[j] * corr + rs;
      mrun[j] = mnew;
#pragma unroll
      for (int t = 0; t < 8; ++t) o[t][j] *= corr;

      const int row = hi * 8 + j;
      myp[row * 32 + ncol]      = f2b(p0);
      myp[row * 32 + 16 + ncol] = f2b(p1);
    }

    asm volatile("s_wait_dscnt 0" ::: "memory");   // per-wave LDS RAW

    // ---- O += P * V  (P: 16x32 A-frag from LDS; V via vT, K-contiguous) ----
    v16bf pf = load_frag_a(myp, 32, 0, lane);
#pragma unroll
    for (int t = 0; t < 8; ++t) {
      v16bf vb = load_frag_b(vtb + (size_t)(t * 16) * SS, SS, key0, lane);
      o[t] = wmma_bf16(pf, vb, o[t]);
    }
  }

  // ---- normalize + store y[b*S+q][h*HD + d] (bf16, row-major [B*S][H]) ----
#pragma unroll
  for (int t = 0; t < 8; ++t)
#pragma unroll
    for (int j = 0; j < 8; ++j) {
      const int qrow = q0 + hi * 8 + j;
      size_t idx = (size_t)(b * SS + qrow) * HH + (size_t)h * HDD + t * 16 + ncol;
      y[idx] = f2b(o[t][j] / lrun[j]);
    }
}

// ---------------- host orchestration ----------------
extern "C" void kernel_launch(void* const* d_in, const int* in_sizes, int n_in,
                              void* d_out, int out_size, void* d_ws, size_t ws_size,
                              hipStream_t stream) {
  const float* x      = (const float*)d_in[0];   // [B,S,H]
  const float* w_attn = (const float*)d_in[1];   // [3H,H]
  const float* w_proj = (const float*)d_in[2];   // [H,H]
  float* out = (float*)d_out;                    // [B,S,H]

  const size_t n_x  = (size_t)BB * SS * HH;      // 8,388,608
  const size_t n_wa = (size_t)H3 * HH;           // 12,582,912
  const size_t n_wp = (size_t)HH * HH;           // 4,194,304
  const size_t n_qkv = (size_t)BB * SS * H3;     // 25,165,824
  const size_t n_vt  = n_x;
  const size_t n_y   = n_x;

  char* ws = (char*)d_ws;
  auto align256 = [](size_t v) { return (v + 255) & ~(size_t)255; };
  size_t off = 0;
  bf16* xb  = (bf16*)(ws + off); off = align256(off + n_x  * sizeof(bf16));
  bf16* wab = (bf16*)(ws + off); off = align256(off + n_wa * sizeof(bf16));
  bf16* wpb = (bf16*)(ws + off); off = align256(off + n_wp * sizeof(bf16));
  bf16* qkv = (bf16*)(ws + off); off = align256(off + n_qkv* sizeof(bf16));
  bf16* vT  = (bf16*)(ws + off); off = align256(off + n_vt * sizeof(bf16));
  bf16* yb  = (bf16*)(ws + off); off = align256(off + n_y  * sizeof(bf16));

  // 1) convert inputs to bf16 (4 elems/thread)
  k_convert4<<<dim3((unsigned)(n_x  / 4 / 256)), 256, 0, stream>>>(x, xb, n_x / 4);
  k_convert4<<<dim3((unsigned)(n_wa / 4 / 256)), 256, 0, stream>>>(w_attn, wab, n_wa / 4);
  k_convert4<<<dim3((unsigned)(n_wp / 4 / 256)), 256, 0, stream>>>(w_proj, wpb, n_wp / 4);

  // 2) QKV projection: [4096,6144] = xb[4096,2048] * wab^T
  k_gemm<bf16><<<dim3(H3 / 128, (BB * SS) / 128), 256, 0, stream>>>(
      xb, wab, qkv, BB * SS, H3, HH);

  // 3) transpose V per head -> vT[b,h,d,s]  (LDS-tiled)
  k_transpose_v<<<dim3(SS / 32, HDD / 32, BB * NHH), 256, 0, stream>>>(qkv, vT);

  // 4) causal flash attention -> yb [B*S, H]
  k_flash_attn<<<dim3(BB * NHH, SS / 128), 256, 0, stream>>>(qkv, vT, yb);

  // 5) output projection: out[4096,2048] = yb * wpb^T  (fp32 out)
  k_gemm<float><<<dim3(HH / 128, (BB * SS) / 128), 256, 0, stream>>>(
      yb, wpb, out, BB * SS, HH, HH);
}
